// LQPolicyModel_49306224558576
// MI455X (gfx1250) — compile-verified, matrix operand order checked by reading, not run on hardware
//
#include <hip/hip_runtime.h>
#include <hip/hip_bf16.h>
#include <math.h>

// ---------------------------------------------------------------------------
// LQ policy rollout, fused persistent kernel for gfx1250 (MI455X).
// BATCH=4096, D=128, DPI=64, DWDIM=128, NH=64, NT=64, HID=256.
// One block = 16 batch rows, 8 wave32s, whole 64-step scan in-block.
// GEMMs via v_wmma_f32_16x16x32_bf16 (fp32 accumulate, fp32 state).
// tanh via native V_TANH_F32 when available.
// ---------------------------------------------------------------------------

typedef __bf16 bf16;
typedef __attribute__((ext_vector_type(16))) __bf16 v16bf;
typedef __attribute__((ext_vector_type(8)))  float  v8f;

#define BATCH_ 4096
#define D_     128
#define DPI_   64
#define NH_    64
#define NT_    64
#define HID_   256
#define DT_    0.05f
#define EXPFAC_ 0.99501247919268232f   // exp(-0.1*DT)

// Swizzled-weight offsets in d_ws (bf16 elements)
#define OFF_A3   0
#define OFF_Wx   16384
#define OFF_Wh   49152
#define OFF_Wo   114688
#define OFF_Q    131072
#define OFF_R    147456
#define OFF_A1T  151552
#define OFF_A2T  167936
#define OFF_A3T  184320
#define OFF_BmT  200704
#define OFF_SgT  208896
#define OFF_G    225280
#define OFF_Wh0  241664

// Native tanh (V_TANH_F32) on gfx1250; libm fallback elsewhere.
__device__ __forceinline__ float fast_tanh(float x) {
#if __has_builtin(__builtin_amdgcn_tanhf)
    return __builtin_amdgcn_tanhf(x);
#elif __has_builtin(__builtin_amdgcn_tanh_f32)
    return __builtin_amdgcn_tanh_f32(x);
#else
    return tanhf(x);
#endif
}

// A-fragment swizzle: element (m, k) -> linear index in per-block A buffer.
// Matches ISA 16-bit A 16x32 layout: lane = m + 16*half, 16 bf16 per lane.
__device__ __forceinline__ int a_swz(int m, int k) {
    int kb   = k >> 5;
    int kk   = k & 31;
    int hi   = kk >> 4;          // K in [16,32) within block
    int k4   = kk & 15;
    int half = k4 >> 3;          // lane half (lanes 16-31 hold K+8)
    int km   = k4 & 7;
    int r    = (hi << 2) + (km >> 1);
    int s    = (r << 1) + (km & 1);
    return ((kb << 5) + (half << 4) + m) * 16 + s;
}

// One 16x16 output tile: C += A(16xK, LDS swizzled) * B(Kx16 col-tile jn,
// global pre-swizzled). K multiple of 32.
__device__ __forceinline__ v8f gemm_tile(const bf16* __restrict__ aL,
                                         const bf16* __restrict__ bS,
                                         int ntiles, int jn, int Ktot,
                                         int lane, v8f c) {
    const int nkb = Ktot >> 5;
    #pragma unroll
    for (int kb = 0; kb < nkb; ++kb) {
        v16bf a = *(const v16bf*)(aL + (((kb << 5) + lane) << 4));
        v16bf b = *(const v16bf*)(bS + ((((kb * ntiles + jn) << 5) + lane) << 4));
        c = __builtin_amdgcn_wmma_f32_16x16x32_bf16(
                false, a, false, b, (short)0, c, false, false);
    }
    return c;
}

// ---------------------------------------------------------------------------
// Prep: convert fp32 weight (K x N effective B-matrix, optionally transposed
// view of src) into bf16, laid out in per-lane WMMA B-fragment order.
// ---------------------------------------------------------------------------
__global__ void swz_kernel(const float* __restrict__ src, bf16* __restrict__ dst,
                           int K, int N, int ld, int transposed) {
    int i = blockIdx.x * 256 + threadIdx.x;
    int total = K * N;
    if (i >= total) return;
    int s    = i & 15;
    int lane = (i >> 4) & 31;
    int rest = i >> 9;
    int ntiles = N >> 4;
    int jn = rest % ntiles;
    int kb = rest / ntiles;
    int half = lane >> 4;
    int n = (jn << 4) | (lane & 15);
    int r = s >> 1;
    int k = (kb << 5)
          + ((r < 4) ? (2 * r + 8 * half) : (16 + 2 * (r - 4) + 8 * half))
          + (s & 1);
    float v = transposed ? src[(size_t)n * ld + k] : src[(size_t)k * ld + n];
    dst[i] = (bf16)v;
}

// ---------------------------------------------------------------------------
// Main fused rollout kernel.
// ---------------------------------------------------------------------------
__global__ __launch_bounds__(256) void lq_rollout(
    const float* __restrict__ dw,     // (B, DWDIM, NT)
    const float* __restrict__ xinit,  // (B, D, NH)
    const float* __restrict__ wgt0,   // (B, D, NH)
    const float* __restrict__ expar,  // (NH,)
    const bf16*  __restrict__ wsz,    // swizzled weights
    float* __restrict__ out)          // (B,)
{
    __shared__ bf16 xA[2][16 * 128];
    __shared__ bf16 hA[2][16 * 256];
    __shared__ bf16 yA[16 * 128];
    __shared__ bf16 zA[16 * 128];
    __shared__ bf16 dwA[16 * 128];
    __shared__ bf16 xcA[16 * 128];
    __shared__ bf16 piA[16 * 64];
    __shared__ float xf[2][16 * 128];
    __shared__ float Sf[16 * 128];
    __shared__ float firstf[16 * 128];
    __shared__ float x0f[16 * 128];
    __shared__ float xcf[16 * 128];
    __shared__ float qxf[16 * 128];
    __shared__ float rrf[16 * 64];
    __shared__ float pif[16 * 64];
    __shared__ float rsum[16];
    __shared__ float racc[16];

    const int tid  = threadIdx.x;
    const int lane = tid & 31;
    const int wave = tid >> 5;
    const int b0   = blockIdx.x * 16;
    const float ea = expar[NH_ - 2];   // exp_array[-2]

    const bf16* W_A3  = wsz + OFF_A3;
    const bf16* W_Wx  = wsz + OFF_Wx;
    const bf16* W_Wh  = wsz + OFF_Wh;
    const bf16* W_Wo  = wsz + OFF_Wo;
    const bf16* W_Q   = wsz + OFF_Q;
    const bf16* W_R   = wsz + OFF_R;
    const bf16* W_A1T = wsz + OFF_A1T;
    const bf16* W_A2T = wsz + OFF_A2T;
    const bf16* W_A3T = wsz + OFF_A3T;
    const bf16* W_BmT = wsz + OFF_BmT;
    const bf16* W_SgT = wsz + OFF_SgT;
    const bf16* W_G   = wsz + OFF_G;
    const bf16* W_Wh0 = wsz + OFF_Wh0;

    // ---- init: x = x_init[:,:,-1]; S = sum(wgt); first0 = wgt[:,:,0] ----
    for (int e = tid; e < 2048; e += 256) {
        int m = e >> 7, d = e & 127;
        size_t base = ((size_t)(b0 + m) * 128 + d) * 64;
        float xv = xinit[base + 63];
        xf[0][e] = xv;
        xA[0][a_swz(m, d)] = (bf16)xv;
        float s = 0.f;
        const float4* wp = (const float4*)(wgt0 + base);
        #pragma unroll
        for (int q = 0; q < 16; ++q) { float4 v = wp[q]; s += v.x + v.y + v.z + v.w; }
        Sf[e] = s;
        firstf[e] = wgt0[base + 0];
    }
    if (tid < 16) racc[tid] = 0.f;
    __syncthreads();

    // ---- hidden = tanh(x @ Wh0), 16 col-tiles, 2 per wave ----
    for (int t = 0; t < 2; ++t) {
        int jn = wave + 8 * t;
        v8f c = {};
        c = gemm_tile(xA[0], W_Wh0, 16, jn, 128, lane, c);
        int half = lane >> 4, nn = (jn << 4) | (lane & 15);
        #pragma unroll
        for (int i = 0; i < 8; ++i)
            hA[0][a_swz(i + 8 * half, nn)] = (bf16)fast_tanh(c[i]);
    }
    __syncthreads();

    int cur = 0;
    float decayN = 1.f;  // ea_m2^n

    for (int n = 0; n < 64; ++n) {
        const int nxt = cur ^ 1;

        // -- shift (n>=1) + stream zeta/first/dw + y --
        for (int e = tid; e < 2048; e += 256) {
            int m = e >> 7, d = e & 127;
            size_t base = ((size_t)(b0 + m) * 128 + d) * 64;
            float xv = xf[cur][e];
            float S = Sf[e], fst, lst;
            if (n == 0) {
                fst = firstf[e];
                lst = wgt0[base + 63];
            } else {
                S = ea * (S - firstf[e]) + xv;
                Sf[e] = S;
                fst = wgt0[base + n] * decayN;
                firstf[e] = fst;
                lst = xv;
            }
            float y = (S - 0.5f * (fst + lst)) * DT_;
            yA[a_swz(m, d)]  = (bf16)y;
            zA[a_swz(m, d)]  = (bf16)xinit[base + n];   // zeta_n
            dwA[a_swz(m, d)] = (bf16)dw[base + n];      // dw[:, :, n]
        }
        __syncthreads();

        // -- x_common = x + EXPFAC*(y@A3)  (8 tiles, 1/wave) --
        {
            v8f c = {};
            c = gemm_tile(yA, W_A3, 8, wave, 128, lane, c);
            int half = lane >> 4, nn = (wave << 4) | (lane & 15);
            #pragma unroll
            for (int i = 0; i < 8; ++i) {
                int m = i + 8 * half, e = (m << 7) | nn;
                float v = xf[cur][e] + EXPFAC_ * c[i];
                xcf[e] = v;
                xcA[a_swz(m, nn)] = (bf16)v;
            }
        }
        // -- h = tanh(x@Wx + hidden@Wh)  (16 tiles, 2/wave) --
        for (int t = 0; t < 2; ++t) {
            int jn = wave + 8 * t;
            v8f c = {};
            c = gemm_tile(xA[cur], W_Wx, 16, jn, 128, lane, c);
            c = gemm_tile(hA[cur], W_Wh, 16, jn, 256, lane, c);
            int half = lane >> 4, nn = (jn << 4) | (lane & 15);
            #pragma unroll
            for (int i = 0; i < 8; ++i)
                hA[nxt][a_swz(i + 8 * half, nn)] = (bf16)fast_tanh(c[i]);
        }
        __syncthreads();

        // -- pi = h @ Wo  (4 tiles, waves 0-3) --
        if (wave < 4) {
            v8f c = {};
            c = gemm_tile(hA[nxt], W_Wo, 4, wave, 256, lane, c);
            int half = lane >> 4, nn = (wave << 4) | (lane & 15);
            #pragma unroll
            for (int i = 0; i < 8; ++i) {
                int m = i + 8 * half, e = (m << 6) | nn;
                pif[e] = c[i];
                piA[a_swz(m, nn)] = (bf16)c[i];
            }
        }
        if (tid < 16) rsum[tid] = 0.f;
        __syncthreads();

        // -- qx = xc@Q (8 tiles) ; rr = pi@R (4 tiles) --
        {
            v8f c = {};
            c = gemm_tile(xcA, W_Q, 8, wave, 128, lane, c);
            int half = lane >> 4, nn = (wave << 4) | (lane & 15);
            #pragma unroll
            for (int i = 0; i < 8; ++i) qxf[((i + 8 * half) << 7) | nn] = c[i];
        }
        if (wave < 4) {
            v8f c = {};
            c = gemm_tile(piA, W_R, 4, wave, 64, lane, c);
            int half = lane >> 4, nn = (wave << 4) | (lane & 15);
            #pragma unroll
            for (int i = 0; i < 8; ++i) rrf[((i + 8 * half) << 6) | nn] = c[i];
        }
        __syncthreads();

        // -- inst_r row reduction --
        {
            int m = tid >> 4, j = tid & 15;
            float p = 0.f;
            #pragma unroll
            for (int q = 0; q < 8; ++q) { int d2 = j + 16 * q; p += qxf[(m << 7) | d2] * xcf[(m << 7) | d2]; }
            #pragma unroll
            for (int q = 0; q < 4; ++q) { int d2 = j + 16 * q; p += rrf[(m << 6) | d2] * pif[(m << 6) | d2]; }
            atomicAdd(&rsum[m], p);
        }
        __syncthreads();
        if (tid < 16) racc[tid] += rsum[tid] * DT_;

        // -- x_new = x + dt*(x@A1T + y@A2T + z@A3T + pi@BmT) + dw@SgT --
        {
            v8f c = {};
            c = gemm_tile(xA[cur], W_A1T, 8, wave, 128, lane, c);
            c = gemm_tile(yA,      W_A2T, 8, wave, 128, lane, c);
            c = gemm_tile(zA,      W_A3T, 8, wave, 128, lane, c);
            c = gemm_tile(piA,     W_BmT, 8, wave, 64,  lane, c);
            v8f c2 = {};
            c2 = gemm_tile(dwA, W_SgT, 8, wave, 128, lane, c2);
            int half = lane >> 4, nn = (wave << 4) | (lane & 15);
            #pragma unroll
            for (int i = 0; i < 8; ++i) {
                int m = i + 8 * half, e = (m << 7) | nn;
                float v = xf[cur][e] + DT_ * c[i] + c2[i];
                xf[nxt][e] = v;
                xA[nxt][a_swz(m, nn)] = (bf16)v;
                if (n == 0) x0f[e] = v;   // needed for first_64
            }
        }
        __syncthreads();

        cur ^= 1;
        decayN *= ea;
    }

    // ---- final shift (64) + terminal reward x_common @ G . x_common ----
    const float decay63 = decayN / ea;  // ea^63
    for (int e = tid; e < 2048; e += 256) {
        int m = e >> 7, d = e & 127;
        float xv = xf[cur][e];
        float S = ea * (Sf[e] - firstf[e]) + xv;
        float fst = x0f[e] * decay63;
        float y = (S - 0.5f * (fst + xv)) * DT_;
        yA[a_swz(m, d)] = (bf16)y;
    }
    __syncthreads();
    {
        v8f c = {};
        c = gemm_tile(yA, W_A3, 8, wave, 128, lane, c);
        int half = lane >> 4, nn = (wave << 4) | (lane & 15);
        #pragma unroll
        for (int i = 0; i < 8; ++i) {
            int m = i + 8 * half, e = (m << 7) | nn;
            float v = xf[cur][e] + EXPFAC_ * c[i];
            xcf[e] = v;
            xcA[a_swz(m, nn)] = (bf16)v;
        }
    }
    if (tid < 16) rsum[tid] = 0.f;
    __syncthreads();
    {
        v8f c = {};
        c = gemm_tile(xcA, W_G, 8, wave, 128, lane, c);
        int half = lane >> 4, nn = (wave << 4) | (lane & 15);
        #pragma unroll
        for (int i = 0; i < 8; ++i) qxf[((i + 8 * half) << 7) | nn] = c[i];
    }
    __syncthreads();
    {
        int m = tid >> 4, j = tid & 15;
        float p = 0.f;
        #pragma unroll
        for (int q = 0; q < 8; ++q) { int d2 = j + 16 * q; p += qxf[(m << 7) | d2] * xcf[(m << 7) | d2]; }
        atomicAdd(&rsum[m], p);
    }
    __syncthreads();
    if (tid < 16) out[b0 + tid] = racc[tid] + rsum[tid];
}

// ---------------------------------------------------------------------------
extern "C" void kernel_launch(void* const* d_in, const int* in_sizes, int n_in,
                              void* d_out, int out_size, void* d_ws, size_t ws_size,
                              hipStream_t stream) {
    const float* dw   = (const float*)d_in[0];
    const float* xin  = (const float*)d_in[1];
    const float* wgt  = (const float*)d_in[2];
    const float* ear  = (const float*)d_in[3];
    const float* A1   = (const float*)d_in[4];
    const float* A2   = (const float*)d_in[5];
    const float* A3   = (const float*)d_in[6];
    const float* Bm   = (const float*)d_in[7];
    const float* Q    = (const float*)d_in[8];
    const float* R    = (const float*)d_in[9];
    const float* G    = (const float*)d_in[10];
    const float* sg   = (const float*)d_in[11];
    const float* Wx   = (const float*)d_in[12];
    const float* Wh   = (const float*)d_in[13];
    const float* Wo   = (const float*)d_in[14];
    const float* Wh0  = (const float*)d_in[15];
    bf16* wsz = (bf16*)d_ws;

    auto prep = [&](const float* src, int off, int K, int N, int ld, int tr) {
        int total = K * N;
        swz_kernel<<<(total + 255) / 256, 256, 0, stream>>>(src, wsz + off, K, N, ld, tr);
    };
    prep(A3,  OFF_A3,  128, 128, 128, 0);
    prep(Wx,  OFF_Wx,  128, 256, 256, 0);
    prep(Wh,  OFF_Wh,  256, 256, 256, 0);
    prep(Wo,  OFF_Wo,  256,  64,  64, 0);
    prep(Q,   OFF_Q,   128, 128, 128, 0);
    prep(R,   OFF_R,    64,  64,  64, 0);
    prep(A1,  OFF_A1T, 128, 128, 128, 1);
    prep(A2,  OFF_A2T, 128, 128, 128, 1);
    prep(A3,  OFF_A3T, 128, 128, 128, 1);
    prep(Bm,  OFF_BmT,  64, 128,  64, 1);
    prep(sg,  OFF_SgT, 128, 128, 128, 1);
    prep(G,   OFF_G,   128, 128, 128, 0);
    prep(Wh0, OFF_Wh0, 128, 256, 256, 0);

    lq_rollout<<<BATCH_ / 16, 256, 0, stream>>>(dw, xin, wgt, ear, wsz, (float*)d_out);
}
